// FocalModulationLayer_21964462751878
// MI455X (gfx1250) — matrix operand order repaired and checked by reading, not compiled, not run on hardware
//
#include <hip/hip_runtime.h>
#include <hip/hip_bf16.h>

typedef __attribute__((ext_vector_type(16))) __bf16 v16bf;
typedef __attribute__((ext_vector_type(8)))  __bf16 v8bf;
typedef __attribute__((ext_vector_type(8)))  float  v8f;

#define NTOK   131072L   // 32*64*64
#define CH     256
#define HW     4096
#define WDIM   64
#define TILE_U 2112      // uints per swizzled weight tile: 16 rows * 132 (stride pad)

// ---- fp32 -> bf16 bits, round-to-nearest-even ----
__device__ __forceinline__ unsigned short bfbits(float f) {
    unsigned u = __builtin_bit_cast(unsigned, f);
    unsigned r = u + 0x7FFFu + ((u >> 16) & 1u);
    return (unsigned short)(r >> 16);
}
__device__ __forceinline__ float bf2f(unsigned short h) {
    return __builtin_bit_cast(float, (unsigned)h << 16);
}
__device__ __forceinline__ float clip100(float v) {
    return fminf(100.f, fmaxf(-100.f, v));
}

// =====================================================================
// Pre-swizzle one weight matrix into per-16-col-tile LDS images:
// out[t][n][kp] = pack_bf16( B[2kp][t*16+n], B[2kp+1][t*16+n] ), kp stride 132.
// One block per tile, 128 threads (one per K-pair).
// =====================================================================
__global__ __launch_bounds__(128) void swizzle_w_kernel(
    const float* __restrict__ B, int ldb, unsigned* __restrict__ outT)
{
    const int t  = blockIdx.x;
    const int kp = threadIdx.x;      // 0..127
    #pragma unroll
    for (int n = 0; n < 16; ++n) {
        float f0 = B[(long)(2 * kp + 0) * ldb + t * 16 + n];
        float f1 = B[(long)(2 * kp + 1) * ldb + t * 16 + n];
        outT[(long)t * TILE_U + n * 132 + kp] =
            (unsigned)bfbits(f0) | ((unsigned)bfbits(f1) << 16);
    }
}

// fp32 tensor -> packed bf16, 8 elements per thread
__global__ __launch_bounds__(256) void cvt_bf16_kernel(
    const float* __restrict__ in, unsigned* __restrict__ out)
{
    const long gid = (long)blockIdx.x * 256 + threadIdx.x;
    const float4* ip = (const float4*)in;
    float4 f0 = ip[2 * gid], f1 = ip[2 * gid + 1];
    uint4 o;
    o.x = (unsigned)bfbits(f0.x) | ((unsigned)bfbits(f0.y) << 16);
    o.y = (unsigned)bfbits(f0.z) | ((unsigned)bfbits(f0.w) << 16);
    o.z = (unsigned)bfbits(f1.x) | ((unsigned)bfbits(f1.y) << 16);
    o.w = (unsigned)bfbits(f1.z) | ((unsigned)bfbits(f1.w) << 16);
    ((uint4*)out)[gid] = o;
}

// =====================================================================
// bf16 WMMA GEMM over pre-swizzled weights.
//   MODE 0: D(f32)  = A*B + bias
//   MODE 1: D(bf16) = clip(A*B + bias) * clip(extra_bf16)   (modulator fuse)
//   MODE 2: D(bf16) = A*B + bias                            (query path)
// Block: 256 thr = 8 waves, each wave one 16x16 tile; block tile 128x16.
// Inner loop: 2x global b128 (A bf16) + 2x ds b128 (B pairs) + 1 WMMA.
// =====================================================================
template <int MODE>
__global__ __launch_bounds__(256) void wmma_gemm_kernel(
    const __bf16* __restrict__ A, int lda,
    const unsigned* __restrict__ Bsw,
    const float* __restrict__ bias,
    void* __restrict__ Dv, int ldc,
    const unsigned short* __restrict__ extra)
{
    __shared__ unsigned lds_b[16 * 132];   // 8448 B

    const int tid  = threadIdx.x;
    const int lane = tid & 31;
    const int wave = tid >> 5;
    const int col  = lane & 15;
    const int half = lane >> 4;
    const int coff = blockIdx.y * 16;

    // stage pre-swizzled tile: pure contiguous b128 copy
    const unsigned* src = Bsw + (long)blockIdx.y * TILE_U;
    for (int i = tid; i < TILE_U / 4; i += 256)
        ((uint4*)lds_b)[i] = ((const uint4*)src)[i];
    __syncthreads();

    const long tile_row = (long)blockIdx.x * 128 + wave * 16;
    const __bf16* __restrict__ Arow = A + (tile_row + col) * (long)lda;

    v8f c = {};
    for (int k0 = 0; k0 < 256; k0 += 32) {
        // A frag: lane = row; VGPR j pair K = half*16 + 2j,2j+1
        // -> one contiguous 16-bf16 run (two 16B loads).
        const __bf16* ap = Arow + k0 + half * 16;
        v8bf alo = *(const v8bf*)ap;
        v8bf ahi = *(const v8bf*)(ap + 8);
        v16bf a = __builtin_shufflevector(alo, ahi,
                    0,1,2,3,4,5,6,7,8,9,10,11,12,13,14,15);

        // B frag: lanes 0-15 -> K 0..15, lanes 16-31 -> K 16..31;
        // VGPR j = pair K = half*16 + 2j,2j+1 at column `col`.
        // LDS image is already in this order: 8 consecutive packed uints.
        const unsigned* bp = lds_b + col * 132 + (k0 >> 1) + half * 8;
        uint4 b0 = *(const uint4*)bp;
        uint4 b1 = *(const uint4*)(bp + 4);
        v8bf blo = __builtin_bit_cast(v8bf, b0);
        v8bf bhi = __builtin_bit_cast(v8bf, b1);
        v16bf b = __builtin_shufflevector(blo, bhi,
                    0,1,2,3,4,5,6,7,8,9,10,11,12,13,14,15);

        c = __builtin_amdgcn_wmma_f32_16x16x32_bf16(
                false, a, false, b, (short)0, c, false, false);
    }

    // epilogue: lane = col, VGPR r -> row half*8 + r
    const float bv = bias[coff + col];
    #pragma unroll
    for (int r = 0; r < 8; ++r) {
        long m = tile_row + half * 8 + r;
        long o = m * (long)ldc + coff + col;
        float val = c[r] + bv;
        if (MODE == 0) {
            ((float*)Dv)[o] = val;
        } else if (MODE == 1) {
            float qv = bf2f(extra[o]);
            ((unsigned short*)Dv)[o] = bfbits(clip100(val) * clip100(qv));
        } else {
            ((unsigned short*)Dv)[o] = bfbits(val);
        }
    }
}

// =====================================================================
// Gate channels: gates[tok][g] = clip(x[tok,:] . w_init[:,512+g] + b, -1, 1)
// =====================================================================
__global__ __launch_bounds__(256) void gates_kernel(
    const float* __restrict__ x, const float* __restrict__ w,
    const float* __restrict__ bias, float* __restrict__ gates)
{
    const int lane = threadIdx.x & 31;
    const long tok = (long)blockIdx.x * 8 + (threadIdx.x >> 5);
    const float* xr = x + tok * CH;
    float s0 = 0.f, s1 = 0.f, s2 = 0.f, s3 = 0.f;
    for (int cix = lane; cix < CH; cix += 32) {
        float xv = xr[cix];
        const float* wr = w + (long)cix * 516;
        s0 += xv * wr[0]; s1 += xv * wr[1];
        s2 += xv * wr[2]; s3 += xv * wr[3];
    }
    #pragma unroll
    for (int off = 16; off > 0; off >>= 1) {
        s0 += __shfl_xor(s0, off, 32);
        s1 += __shfl_xor(s1, off, 32);
        s2 += __shfl_xor(s2, off, 32);
        s3 += __shfl_xor(s3, off, 32);
    }
    if (lane == 0) {
        float* g = gates + tok * 4;
        g[0] = fminf(1.f, fmaxf(-1.f, s0 + bias[0]));
        g[1] = fminf(1.f, fmaxf(-1.f, s1 + bias[1]));
        g[2] = fminf(1.f, fmaxf(-1.f, s2 + bias[2]));
        g[3] = fminf(1.f, fmaxf(-1.f, s3 + bias[3]));
    }
}

// =====================================================================
// Depthwise conv (same pad) + exact GELU + gated accumulate (NHWC).
// =====================================================================
__global__ __launch_bounds__(256) void dwconv_kernel(
    const float* __restrict__ in, const float* __restrict__ kern, int ks,
    const float* __restrict__ gates, int level,
    float* __restrict__ ctx_out, float* __restrict__ acc, int initFlag)
{
    const long idx = blockIdx.x;
    const int  cix = threadIdx.x;
    const int  hw  = (int)(idx & (HW - 1));
    const long bb  = idx >> 12;
    const int  h   = hw >> 6;
    const int  w   = hw & 63;
    const int  p   = ks >> 1;

    float s = 0.f;
    for (int dh = 0; dh < ks; ++dh) {
        int hh = h + dh - p;
        if ((unsigned)hh >= (unsigned)WDIM) continue;
        for (int dw = 0; dw < ks; ++dw) {
            int ww = w + dw - p;
            if ((unsigned)ww >= (unsigned)WDIM) continue;
            s += in[((bb * HW + hh * WDIM + ww) * CH) + cix]
               * kern[(dh * ks + dw) * CH + cix];
        }
    }
    float g = 0.5f * s * (1.f + erff(s * 0.70710678118654752f));
    long o = idx * CH + cix;
    ctx_out[o] = g;
    float gt = gates[idx * 4 + level];
    if (initFlag) acc[o] = g * gt;
    else          acc[o] += g * gt;
}

__global__ __launch_bounds__(256) void mean_tanh_kernel(
    const float* __restrict__ ctx, float* __restrict__ cg)
{
    const int b = blockIdx.x, cix = threadIdx.x;
    const float* p = ctx + (long)b * HW * CH + cix;
    float s = 0.f;
    for (int hw = 0; hw < HW; ++hw) s += p[(long)hw * CH];
    cg[b * CH + cix] = tanhf(s * (1.f / 4096.f));
}

// accbf16 = bf16(acc + cg[b,c] * gates[tok,3])   (emits GEMM-ready bf16 A)
__global__ __launch_bounds__(256) void add_global_bf16_kernel(
    const float* __restrict__ acc, const float* __restrict__ cg,
    const float* __restrict__ gates, unsigned short* __restrict__ out)
{
    const long tok = blockIdx.x;
    const int  cix = threadIdx.x;
    const long bb  = tok >> 12;
    float v = acc[tok * CH + cix] + cg[bb * CH + cix] * gates[tok * 4 + 3];
    out[tok * CH + cix] = bfbits(v);
}

extern "C" void kernel_launch(void* const* d_in, const int* in_sizes, int n_in,
                              void* d_out, int out_size, void* d_ws, size_t ws_size,
                              hipStream_t stream) {
    const float* x      = (const float*)d_in[0];
    const float* w_init = (const float*)d_in[1];
    const float* b_init = (const float*)d_in[2];
    const float* k0     = (const float*)d_in[3];
    const float* k1     = (const float*)d_in[4];
    const float* k2     = (const float*)d_in[5];
    const float* w_mod  = (const float*)d_in[6];
    const float* b_mod  = (const float*)d_in[7];
    const float* w_proj = (const float*)d_in[8];
    const float* b_proj = (const float*)d_in[9];
    float* out = (float*)d_out;

    float* ws   = (float*)d_ws;
    // f32 regions
    float* qbfR = ws;                         // N x 256 bf16  (N*128 floats)
    float* ctxA = ws + NTOK * 128;            // N x 256 f32
    float* ctxB = ctxA + NTOK * CH;           // N x 256 f32
    float* acc  = ctxB + NTOK * CH;           // N x 256 f32
    float* xbfR = acc  + NTOK * CH;           // N x 256 bf16  (N*128 floats)
    float* gat  = xbfR + NTOK * 128;          // N x 4
    float* cg   = gat + NTOK * 4;             // 32 x 256
    unsigned* wsw = (unsigned*)(cg + 32 * CH);// 64 tiles x 2112 uints
    unsigned* wsw_init = wsw;                 // 32 tiles (w_init cols 0..511)
    unsigned* wsw_mod  = wsw + 32L * TILE_U;  // 16 tiles
    unsigned* wsw_proj = wsw + 48L * TILE_U;  // 16 tiles

    // bf16 views (aliases: free after their f32 producers are consumed)
    unsigned short* qbf   = (unsigned short*)qbfR;
    unsigned short* xbf   = (unsigned short*)xbfR;
    unsigned short* accbf = (unsigned short*)ctxB;  // alias: ctxB dead after mean
    unsigned short* tbf   = (unsigned short*)ctxA;  // alias: ctxA dead after conv2

    dim3 blk(256);
    dim3 ggrid((unsigned)(NTOK / 128), CH / 16);

    // ---- prep: activations -> bf16, weights -> swizzled bf16 tiles ----
    cvt_bf16_kernel<<<(unsigned)(NTOK * CH / (256 * 8)), blk, 0, stream>>>(
        x, (unsigned*)xbf);
    swizzle_w_kernel<<<32, 128, 0, stream>>>(w_init, 516, wsw_init);
    swizzle_w_kernel<<<16, 128, 0, stream>>>(w_mod, CH, wsw_mod);
    swizzle_w_kernel<<<16, 128, 0, stream>>>(w_proj, CH, wsw_proj);

    // ---- x_proj: query (bf16), context (f32), gates ----
    wmma_gemm_kernel<2><<<ggrid, blk, 0, stream>>>(
        (const __bf16*)xbf, CH, wsw_init, b_init, qbf, CH, nullptr);
    wmma_gemm_kernel<0><<<ggrid, blk, 0, stream>>>(
        (const __bf16*)xbf, CH, wsw_init + 16L * TILE_U, b_init + 256,
        ctxA, CH, nullptr);
    gates_kernel<<<(unsigned)(NTOK / 8), blk, 0, stream>>>(
        x, w_init + 512, b_init + 512, gat);

    // ---- focal levels: 3x3 -> 5x5 -> 7x7 (ping-pong) ----
    dwconv_kernel<<<(unsigned)NTOK, blk, 0, stream>>>(ctxA, k0, 3, gat, 0, ctxB, acc, 1);
    dwconv_kernel<<<(unsigned)NTOK, blk, 0, stream>>>(ctxB, k1, 5, gat, 1, ctxA, acc, 0);
    dwconv_kernel<<<(unsigned)NTOK, blk, 0, stream>>>(ctxA, k2, 7, gat, 2, ctxB, acc, 0);

    // ---- global context + gated add, emitted as bf16 GEMM operand ----
    mean_tanh_kernel<<<32, blk, 0, stream>>>(ctxB, cg);
    add_global_bf16_kernel<<<(unsigned)NTOK, blk, 0, stream>>>(acc, cg, gat, accbf);

    // ---- t = clip(q)*clip(acc@w_mod + b_mod) -> bf16; out = t@w_proj + b ----
    wmma_gemm_kernel<1><<<ggrid, blk, 0, stream>>>(
        (const __bf16*)accbf, CH, wsw_mod, b_mod, tbf, CH, qbf);
    wmma_gemm_kernel<0><<<ggrid, blk, 0, stream>>>(
        (const __bf16*)tbf, CH, wsw_proj, b_proj, out, CH, nullptr);
}
